// LocalWindowTransformer_81716047773734
// MI455X (gfx1250) — matrix operand order repaired and checked by reading, not compiled, not run on hardware
//
#include <hip/hip_runtime.h>
#include <hip/hip_bf16.h>
#include <cstddef>

// ---------------- problem constants (match reference) ----------------
#define B_  8
#define S_  1024
#define D_  1024
#define H_  8
#define HD_ 128
#define L_  4
#define FF_ 4096
#define NOUT_ 512
#define M_  (B_ * S_)   // 8192 rows of activations

typedef __bf16 bf16_t;
typedef bf16_t bf16x16 __attribute__((ext_vector_type(16)));
typedef float  f32x8   __attribute__((ext_vector_type(8)));

// Build a 16x32 A-fragment for one lane: halves {akb..akb+7, 16+akb..16+akb+7}
__device__ __forceinline__ bf16x16 make_afrag(const float* __restrict__ p, int akb)
{
    const float4 x0 = *(const float4*)(p + akb);
    const float4 x1 = *(const float4*)(p + akb + 4);
    const float4 x2 = *(const float4*)(p + akb + 16);
    const float4 x3 = *(const float4*)(p + akb + 20);
    bf16x16 a;
    a[0]=(bf16_t)x0.x; a[1]=(bf16_t)x0.y; a[2]=(bf16_t)x0.z; a[3]=(bf16_t)x0.w;
    a[4]=(bf16_t)x1.x; a[5]=(bf16_t)x1.y; a[6]=(bf16_t)x1.z; a[7]=(bf16_t)x1.w;
    a[8]=(bf16_t)x2.x; a[9]=(bf16_t)x2.y; a[10]=(bf16_t)x2.z; a[11]=(bf16_t)x2.w;
    a[12]=(bf16_t)x3.x; a[13]=(bf16_t)x3.y; a[14]=(bf16_t)x3.z; a[15]=(bf16_t)x3.w;
    return a;
}

// =====================================================================
// GEMM: out[M,N] = act( A[M,K] * W[N,K]^T + bias[N] (+ resid[M,N]) )
// bf16 WMMA 16x16x32, f32 accumulate. Block = 128 threads (4 waves).
// Wave tile = 32(M) x 64(N): 2 A-frags x 4 B-frags = 8 WMMAs / K-slab.
// B tile (64N x 32K bf16) double-buffered through LDS.
// =====================================================================
template<bool RES, bool RELU>
__global__ __launch_bounds__(128) void gemm_bf16_wmma(
    const float* __restrict__ A, const float* __restrict__ W,
    const float* __restrict__ bias, const float* __restrict__ resid,
    float* __restrict__ out, int M, int N, int K)
{
    // rows padded to 40 halves (80B): 16B-aligned rows, 20-dword bank stride
    __shared__ bf16_t Bs[2][64][40];

    const int tid  = threadIdx.x;
    const int wave = tid >> 5;
    const int lane = tid & 31;
    const int lo   = lane & 15;
    const bool hi  = lane >= 16;

    const int m0  = blockIdx.y * 128 + wave * 32;
    const int n0  = blockIdx.x * 64;
    const int akb = hi ? 8 : 0;       // A K sub-base
    const int bkb = hi ? 8 : 0;       // B K sub-base within 32-K slab

    // cooperative B-loader assignment: 128 threads x 16 f32 = 64 x 32 tile
    const int ldn  = tid >> 1;
    const int ldks = (tid & 1) * 16;
    const float* wsrc = W + (size_t)(n0 + ldn) * K + ldks;

    f32x8 acc0[4], acc1[4];
#pragma unroll
    for (int t = 0; t < 4; ++t) {
        acc0[t] = (f32x8){0.f,0.f,0.f,0.f,0.f,0.f,0.f,0.f};
        acc1[t] = (f32x8){0.f,0.f,0.f,0.f,0.f,0.f,0.f,0.f};
    }

    // ---- preload slab 0 into Bs[0] ----
    {
        const float4 w0 = *(const float4*)(wsrc + 0);
        const float4 w1 = *(const float4*)(wsrc + 4);
        const float4 w2 = *(const float4*)(wsrc + 8);
        const float4 w3 = *(const float4*)(wsrc + 12);
        bf16_t* dst = &Bs[0][ldn][ldks];
        dst[0]=(bf16_t)w0.x; dst[1]=(bf16_t)w0.y; dst[2]=(bf16_t)w0.z; dst[3]=(bf16_t)w0.w;
        dst[4]=(bf16_t)w1.x; dst[5]=(bf16_t)w1.y; dst[6]=(bf16_t)w1.z; dst[7]=(bf16_t)w1.w;
        dst[8]=(bf16_t)w2.x; dst[9]=(bf16_t)w2.y; dst[10]=(bf16_t)w2.z; dst[11]=(bf16_t)w2.w;
        dst[12]=(bf16_t)w3.x; dst[13]=(bf16_t)w3.y; dst[14]=(bf16_t)w3.z; dst[15]=(bf16_t)w3.w;
    }
    __syncthreads();

    const float* arow0 = A + (size_t)(m0 + lo)      * K;
    const float* arow1 = A + (size_t)(m0 + 16 + lo) * K;

    const int nslab = K >> 5;
    for (int s = 0; s < nslab; ++s) {
        const int buf = s & 1;
        const int k0  = s << 5;
        const bool more = (s + 1) < nslab;

        // ---- issue global loads for next slab early (latency hiding) ----
        float4 w0, w1, w2, w3;
        if (more) {
            const float* ns = wsrc + k0 + 32;
            w0 = *(const float4*)(ns + 0);
            w1 = *(const float4*)(ns + 4);
            w2 = *(const float4*)(ns + 8);
            w3 = *(const float4*)(ns + 12);
            __builtin_prefetch(ns + 32, 0, 0);   // slab s+2 -> global_prefetch_b8
        }

        // ---- A fragments for the two 16-row halves ----
        const bf16x16 a0 = make_afrag(arow0 + k0, akb);
        const bf16x16 a1 = make_afrag(arow1 + k0, akb);

        // ---- pull all 4 B fragments out of LDS before the WMMA burst ----
        bf16x16 b[4];
#pragma unroll
        for (int t = 0; t < 4; ++t) {
            const bf16_t* brow = &Bs[buf][t * 16 + lo][bkb];
#pragma unroll
            for (int i = 0; i < 16; ++i) b[t][i] = brow[i < 8 ? i : i + 8];
        }

        // ---- 8 back-to-back WMMAs ----
#pragma unroll
        for (int t = 0; t < 4; ++t) {
            acc0[t] = __builtin_amdgcn_wmma_f32_16x16x32_bf16(
                false, a0, false, b[t], (short)0, acc0[t], false, false);
            acc1[t] = __builtin_amdgcn_wmma_f32_16x16x32_bf16(
                false, a1, false, b[t], (short)0, acc1[t], false, false);
        }

        // ---- convert + store next slab into the other LDS buffer ----
        if (more) {
            bf16_t* dst = &Bs[buf ^ 1][ldn][ldks];
            dst[0]=(bf16_t)w0.x; dst[1]=(bf16_t)w0.y; dst[2]=(bf16_t)w0.z; dst[3]=(bf16_t)w0.w;
            dst[4]=(bf16_t)w1.x; dst[5]=(bf16_t)w1.y; dst[6]=(bf16_t)w1.z; dst[7]=(bf16_t)w1.w;
            dst[8]=(bf16_t)w2.x; dst[9]=(bf16_t)w2.y; dst[10]=(bf16_t)w2.z; dst[11]=(bf16_t)w2.w;
            dst[12]=(bf16_t)w3.x; dst[13]=(bf16_t)w3.y; dst[14]=(bf16_t)w3.z; dst[15]=(bf16_t)w3.w;
        }
        __syncthreads();
    }

    // ---- epilogue: bias (+residual) (+relu), branch-free ----
#pragma unroll
    for (int t = 0; t < 4; ++t) {
        const int n  = n0 + t * 16 + lo;
        const float bv = bias[n];
#pragma unroll
        for (int r = 0; r < 8; ++r) {
            const int ma = m0 + r + (hi ? 8 : 0);
            const int mb = ma + 16;
            float va = acc0[t][r] + bv;
            float vb = acc1[t][r] + bv;
            if (RES) {
                va += resid[(size_t)ma * N + n];
                vb += resid[(size_t)mb * N + n];
            }
            if (RELU) { va = fmaxf(va, 0.f); vb = fmaxf(vb, 0.f); }
            out[(size_t)ma * N + n] = va;
            out[(size_t)mb * N + n] = vb;
        }
    }
}

// =====================================================================
// x = X + sinusoidal positional encoding
// =====================================================================
__global__ __launch_bounds__(256) void posenc_kernel(const float* __restrict__ X,
                                                     float* __restrict__ x)
{
    const size_t idx = (size_t)blockIdx.x * 256 + threadIdx.x;
    const int d = (int)(idx % D_);
    const int s = (int)((idx / D_) % S_);
    const int i2 = d & ~1;
    const float dv = __expf((float)i2 * (-9.210340371976184f / (float)D_));
    const float ang = (float)s * dv;
    const float pe  = (d & 1) ? __cosf(ang) : __sinf(ang);
    x[idx] = X[idx] + pe;
}

// =====================================================================
// Windowed causal attention. grid=(S/32, H, B), block=1024 (32 waves).
// One wave per query; lanes = 32 window offsets; shuffle softmax;
// probabilities through LDS; lanes switch to d-parallel for P*V.
// =====================================================================
__global__ __launch_bounds__(1024) void attn_kernel(
    const float* __restrict__ q, const float* __restrict__ k,
    const float* __restrict__ v, float* __restrict__ o)
{
    __shared__ float sP[32][33];

    const int wave = threadIdx.x >> 5;
    const int lane = threadIdx.x & 31;
    const int i = blockIdx.x * 32 + wave;
    const int h = blockIdx.y;
    const int b = blockIdx.z;

    const size_t base = ((size_t)b * S_) * D_ + (size_t)h * HD_;
    const float* qrow = q + base + (size_t)i * D_;

    const int j = i - 31 + lane;
    float score = -1e30f;
    if (j >= 0) {
        const float* krow = k + base + (size_t)j * D_;
        float acc = 0.f;
#pragma unroll
        for (int d = 0; d < HD_; d += 4) {
            float4 qv = *(const float4*)(qrow + d);
            float4 kv = *(const float4*)(krow + d);
            acc = fmaf(qv.x, kv.x, fmaf(qv.y, kv.y,
                  fmaf(qv.z, kv.z, fmaf(qv.w, kv.w, acc))));
        }
        score = acc * 0.08838834764831845f;   // 1/sqrt(128)
    }

    float mx = score;
#pragma unroll
    for (int off = 16; off; off >>= 1) mx = fmaxf(mx, __shfl_xor(mx, off, 32));
    float p = (j >= 0) ? __expf(score - mx) : 0.f;
    float sum = p;
#pragma unroll
    for (int off = 16; off; off >>= 1) sum += __shfl_xor(sum, off, 32);
    p /= sum;

    sP[wave][lane] = p;
    __syncthreads();

    const int d0 = lane * 4;
    float4 accv = {0.f, 0.f, 0.f, 0.f};
#pragma unroll
    for (int l = 0; l < 32; ++l) {
        const int jj = i - 31 + l;
        if (jj < 0) continue;
        const float pl = sP[wave][l];
        float4 vv = *(const float4*)(v + base + (size_t)jj * D_ + d0);
        accv.x = fmaf(pl, vv.x, accv.x);
        accv.y = fmaf(pl, vv.y, accv.y);
        accv.z = fmaf(pl, vv.z, accv.z);
        accv.w = fmaf(pl, vv.w, accv.w);
    }
    *(float4*)(o + base + (size_t)i * D_ + d0) = accv;
}

// =====================================================================
// LayerNorm: x = (y - mean)/sqrt(var+eps) * g + b ; one block per row.
// =====================================================================
__global__ __launch_bounds__(256) void layernorm_kernel(
    const float* __restrict__ y, const float* __restrict__ g,
    const float* __restrict__ b, float* __restrict__ x)
{
    const size_t row = blockIdx.x;
    const float* yr = y + row * (size_t)D_;
    float s = 0.f, ss = 0.f;
    for (int i = threadIdx.x; i < D_; i += 256) {
        const float v = yr[i];
        s += v; ss += v * v;
    }
#pragma unroll
    for (int off = 16; off; off >>= 1) {
        s  += __shfl_xor(s,  off, 32);
        ss += __shfl_xor(ss, off, 32);
    }
    __shared__ float rS[8], rSS[8];
    if ((threadIdx.x & 31) == 0) {
        rS[threadIdx.x >> 5]  = s;
        rSS[threadIdx.x >> 5] = ss;
    }
    __syncthreads();
    s = 0.f; ss = 0.f;
#pragma unroll
    for (int w = 0; w < 8; ++w) { s += rS[w]; ss += rSS[w]; }
    const float mean = s * (1.0f / D_);
    const float var  = ss * (1.0f / D_) - mean * mean;
    const float inv  = rsqrtf(var + 1e-5f);
    float* xr = x + row * (size_t)D_;
    for (int i = threadIdx.x; i < D_; i += 256)
        xr[i] = (yr[i] - mean) * inv * g[i] + b[i];
}

// =====================================================================
// host-side orchestration
// =====================================================================
static inline void launch_gemm(const float* A, const float* W, const float* bias,
                               const float* resid, float* out, int M, int N, int K,
                               bool relu, hipStream_t stream)
{
    const dim3 g(N / 64, M / 128), blk(128);
    if (resid)
        gemm_bf16_wmma<true,  false><<<g, blk, 0, stream>>>(A, W, bias, resid, out, M, N, K);
    else if (relu)
        gemm_bf16_wmma<false, true ><<<g, blk, 0, stream>>>(A, W, bias, nullptr, out, M, N, K);
    else
        gemm_bf16_wmma<false, false><<<g, blk, 0, stream>>>(A, W, bias, nullptr, out, M, N, K);
}

extern "C" void kernel_launch(void* const* d_in, const int* in_sizes, int n_in,
                              void* d_out, int out_size, void* d_ws, size_t ws_size,
                              hipStream_t stream)
{
    (void)in_sizes; (void)n_in; (void)out_size; (void)ws_size;
    const float* X    = (const float*)d_in[0];
    const float* Wq   = (const float*)d_in[1];
    const float* Wk   = (const float*)d_in[2];
    const float* Wv   = (const float*)d_in[3];
    const float* bq   = (const float*)d_in[4];
    const float* bk   = (const float*)d_in[5];
    const float* bv   = (const float*)d_in[6];
    const float* Wo   = (const float*)d_in[7];
    const float* bo   = (const float*)d_in[8];
    const float* ln1g = (const float*)d_in[9];
    const float* ln1b = (const float*)d_in[10];
    const float* W1   = (const float*)d_in[11];
    const float* b1   = (const float*)d_in[12];
    const float* W2   = (const float*)d_in[13];
    const float* b2   = (const float*)d_in[14];
    const float* ln2g = (const float*)d_in[15];
    const float* ln2b = (const float*)d_in[16];
    const float* Wh   = (const float*)d_in[17];
    const float* bh   = (const float*)d_in[18];
    float* out = (float*)d_out;

    const size_t BSD = (size_t)M_ * D_;
    float* x  = (float*)d_ws;
    float* q  = x  + BSD;
    float* kk = q  + BSD;
    float* v  = kk + BSD;
    float* o  = v  + BSD;
    float* y  = o  + BSD;
    float* ff = y  + BSD;                 // M_*FF_ floats

    posenc_kernel<<<(unsigned)(BSD / 256), 256, 0, stream>>>(X, x);

    for (int l = 0; l < L_; ++l) {
        const size_t wDD = (size_t)l * D_ * D_;
        const size_t wFD = (size_t)l * FF_ * D_;
        launch_gemm(x, Wq + wDD, bq + (size_t)l*D_, nullptr, q,  M_, D_, D_, false, stream);
        launch_gemm(x, Wk + wDD, bk + (size_t)l*D_, nullptr, kk, M_, D_, D_, false, stream);
        launch_gemm(x, Wv + wDD, bv + (size_t)l*D_, nullptr, v,  M_, D_, D_, false, stream);
        attn_kernel<<<dim3(S_ / 32, H_, B_), 1024, 0, stream>>>(q, kk, v, o);
        launch_gemm(o, Wo + wDD, bo + (size_t)l*D_, x, y, M_, D_, D_, false, stream);
        layernorm_kernel<<<M_, 256, 0, stream>>>(y, ln1g + (size_t)l*D_, ln1b + (size_t)l*D_, x);
        launch_gemm(x,  W1 + wFD, b1 + (size_t)l*FF_, nullptr, ff, M_, FF_, D_, true, stream);
        launch_gemm(ff, W2 + wFD, b2 + (size_t)l*D_,  x,       y,  M_, D_, FF_, false, stream);
        layernorm_kernel<<<M_, 256, 0, stream>>>(y, ln2g + (size_t)l*D_, ln2b + (size_t)l*D_, x);
    }

    launch_gemm(x, Wh, bh, nullptr, out, M_, NOUT_, D_, false, stream);
}
